// GetModel_59176059404796
// MI455X (gfx1250) — matrix operand order repaired and checked by reading, not compiled, not run on hardware
//
#include <hip/hip_runtime.h>
#include <hip/hip_bf16.h>

#define NPOINT   512
#define NSAMPLE  32
#define RADIUS2  0.04f
#define BATCH    16
#define NPTS     4096
#define BN_EPS   1e-5f

typedef __attribute__((ext_vector_type(16))) __bf16 v16bf;
typedef __attribute__((ext_vector_type(8)))  __bf16 v8bf;
typedef __attribute__((ext_vector_type(8)))  float  v8f;

#define GLOBAL_AS __attribute__((address_space(1)))
#define LDS_AS    __attribute__((address_space(3)))

// 16-byte GCC-vector type matching the async-to-LDS builtin's parameter type
typedef int async_i4 __attribute__((vector_size(16)));

#if __has_builtin(__builtin_amdgcn_global_load_async_to_lds_b128) && \
    __has_builtin(__builtin_amdgcn_s_wait_asynccnt)
#define USE_ASYNC_LDS 1
#else
#define USE_ASYNC_LDS 0
#endif

// ---------------------------------------------------------------------------
// 1) Extract p (B,N,3) and norm (B,N,3) from xyz (B,6,N)
// ---------------------------------------------------------------------------
__global__ void extract_kernel(const float* __restrict__ xyz,
                               float* __restrict__ p, float* __restrict__ nrm) {
    int t = blockIdx.x * 256 + threadIdx.x;          // 16*4096 = 65536
    if (t >= BATCH * NPTS) return;
    int b = t >> 12, n = t & (NPTS - 1);
    size_t base = (size_t)b * 6 * NPTS + n;
#pragma unroll
    for (int k = 0; k < 3; ++k) {
        p  [(size_t)t * 3 + k] = xyz[base + (size_t)k * NPTS];
        nrm[(size_t)t * 3 + k] = xyz[base + (size_t)(3 + k) * NPTS];
    }
}

// ---------------------------------------------------------------------------
// 2) Farthest-point sampling: one workgroup per batch, dist[] lives in LDS
// ---------------------------------------------------------------------------
__global__ void __launch_bounds__(256)
fps_kernel(const float* __restrict__ p, int* __restrict__ idx) {
    __shared__ float dist[NPTS];
    __shared__ float rmax[256];
    __shared__ int   rarg[256];
    __shared__ float fpt[3];
    int b = blockIdx.x, tid = threadIdx.x;
    const float* pb = p + (size_t)b * NPTS * 3;
    for (int i = tid; i < NPTS; i += 256) dist[i] = 1e10f;
    int far = 0;
    __syncthreads();
    for (int s = 0; s < NPOINT; ++s) {
        if (tid == 0) {
            idx[b * NPOINT + s] = far;
            fpt[0] = pb[far * 3 + 0];
            fpt[1] = pb[far * 3 + 1];
            fpt[2] = pb[far * 3 + 2];
        }
        __syncthreads();
        float fx = fpt[0], fy = fpt[1], fz = fpt[2];
        float bmax = -1.0f; int barg = 0;
        for (int i = tid; i < NPTS; i += 256) {
            float dx = pb[i * 3 + 0] - fx;
            float dy = pb[i * 3 + 1] - fy;
            float dz = pb[i * 3 + 2] - fz;
            float d  = dx * dx + dy * dy + dz * dz;
            float nd = fminf(dist[i], d);
            dist[i] = nd;
            if (nd > bmax) { bmax = nd; barg = i; }
        }
        rmax[tid] = bmax; rarg[tid] = barg;
        __syncthreads();
        for (int off = 128; off > 0; off >>= 1) {
            if (tid < off && rmax[tid + off] > rmax[tid]) {
                rmax[tid] = rmax[tid + off];
                rarg[tid] = rarg[tid + off];
            }
            __syncthreads();
        }
        far = rarg[0];
        __syncthreads();
    }
}

// ---------------------------------------------------------------------------
// 3) Ball query + grouping: one wave32 per (b,s). Collect 32 smallest in-radius
//    indices in order via ballot + prefix popcount; replicate first if short.
// ---------------------------------------------------------------------------
__global__ void __launch_bounds__(256)
ballquery_kernel(const float* __restrict__ p, const float* __restrict__ nrm,
                 const int* __restrict__ idx, float* __restrict__ new_xyz,
                 float* __restrict__ h0) {
    __shared__ int sidx[8][NSAMPLE];
    int gthread = blockIdx.x * 256 + threadIdx.x;
    int wave = gthread >> 5;                      // 0..8191
    int lane = threadIdx.x & 31;
    int lw   = threadIdx.x >> 5;
    int b = wave / NPOINT, s = wave % NPOINT;
    const float* pb = p   + (size_t)b * NPTS * 3;
    const float* nb = nrm + (size_t)b * NPTS * 3;
    int center = idx[b * NPOINT + s];
    float qx = pb[center * 3 + 0], qy = pb[center * 3 + 1], qz = pb[center * 3 + 2];
    if (lane < 3) new_xyz[(size_t)wave * 3 + lane] = pb[center * 3 + lane];

    int cnt = 0;
    for (int base = 0; base < NPTS && cnt < NSAMPLE; base += 32) {
        int i = base + lane;
        float dx = pb[i * 3 + 0] - qx;
        float dy = pb[i * 3 + 1] - qy;
        float dz = pb[i * 3 + 2] - qz;
        bool in = (dx * dx + dy * dy + dz * dz) <= RADIUS2;
#if __has_builtin(__builtin_amdgcn_ballot_w32)
        unsigned mask = __builtin_amdgcn_ballot_w32(in);
#else
        unsigned mask = (unsigned)__ballot(in);
#endif
        int rank = __popc(mask & ((1u << lane) - 1u));
        if (in && (cnt + rank) < NSAMPLE) sidx[lw][cnt + rank] = i;
        cnt += __popc(mask);
        if (cnt > NSAMPLE) cnt = NSAMPLE;
    }
    if (cnt == 0) { if (lane == 0) sidx[lw][0] = center; cnt = 1; }
    __syncthreads();
    int first = sidx[lw][0];
    int gi = first;
    if (lane < cnt) gi = sidx[lw][lane];
    size_t row = ((size_t)wave * NSAMPLE + lane) * 6;
    h0[row + 0] = pb[gi * 3 + 0] - qx;
    h0[row + 1] = pb[gi * 3 + 1] - qy;
    h0[row + 2] = pb[gi * 3 + 2] - qz;
    h0[row + 3] = nb[gi * 3 + 0];
    h0[row + 4] = nb[gi * 3 + 1];
    h0[row + 5] = nb[gi * 3 + 2];
}

// ---------------------------------------------------------------------------
// 4) Layer 1 GEMM: (262144 x 6) @ (6 x 256) + b. K too small for WMMA -> FMA.
// ---------------------------------------------------------------------------
__global__ void __launch_bounds__(256)
gemm1_kernel(const float* __restrict__ h0, const float* __restrict__ w1,
             const float* __restrict__ b1, float* __restrict__ pre) {
    __shared__ float ws[6 * 256];
    for (int i = threadIdx.x; i < 6 * 256; i += 256) ws[i] = w1[i];
    __syncthreads();
    size_t row = blockIdx.x;
    int c = threadIdx.x;
    const float* a = h0 + row * 6;
    float acc = b1[c];
#pragma unroll
    for (int k = 0; k < 6; ++k) acc += a[k] * ws[k * 256 + c];
    pre[row * 256 + c] = acc;
}

// ---------------------------------------------------------------------------
// 5) Deterministic BN statistics: per-block partials then fixed-order reduce.
// ---------------------------------------------------------------------------
__global__ void stats_partial_kernel(const float* __restrict__ src, long long rows,
                                     int C, int nblk, float* __restrict__ psum,
                                     float* __restrict__ psq) {
    int c = threadIdx.x, blk = blockIdx.x;
    long long per = (rows + nblk - 1) / nblk;
    long long r0 = (long long)blk * per;
    long long r1 = r0 + per; if (r1 > rows) r1 = rows;
    float s = 0.f, q = 0.f;
    for (long long r = r0; r < r1; ++r) {
        float v = src[r * C + c];
        s += v; q += v * v;
    }
    psum[(size_t)blk * C + c] = s;
    psq [(size_t)blk * C + c] = q;
}

__global__ void stats_final_kernel(const float* __restrict__ psum,
                                   const float* __restrict__ psq, int C, int nblk,
                                   float* __restrict__ sum, float* __restrict__ sq) {
    int c = blockIdx.x * blockDim.x + threadIdx.x;
    if (c >= C) return;
    float s = 0.f, q = 0.f;
    for (int b = 0; b < nblk; ++b) {
        s += psum[(size_t)b * C + c];
        q += psq [(size_t)b * C + c];
    }
    sum[c] = s; sq[c] = q;
}

// ---------------------------------------------------------------------------
// 6) BN + ReLU, optionally converting to bf16 for the next WMMA GEMM
// ---------------------------------------------------------------------------
template <bool BF16OUT>
__global__ void bnrelu_kernel(const float* __restrict__ pre,
                              const float* __restrict__ sum, const float* __restrict__ sq,
                              const float* __restrict__ gamma, const float* __restrict__ beta,
                              float invN, void* __restrict__ out, long long total, int cmask) {
    long long i = (long long)blockIdx.x * 256 + threadIdx.x;
    if (i >= total) return;
    int c = (int)(i & cmask);
    float m = sum[c] * invN;
    float v = sq[c] * invN - m * m;
    float x = gamma[c] * (pre[i] - m) * rsqrtf(v + BN_EPS) + beta[c];
    x = fmaxf(x, 0.f);
    if (BF16OUT) ((__hip_bfloat16*)out)[i] = __float2bfloat16(x);
    else         ((float*)out)[i] = x;
}

// ---------------------------------------------------------------------------
// 7) Pack 256x256 f32 weight into kstep-major WMMA bf16 B-fragment layout:
//    wp[((kstep*16 + ntile)*32 + lane)*16 + j] = w[k*256 + n]
//    n = ntile*16 + lane%16, k = kstep*32 + (lane/16)*8 + (j%8) + (j>=8 ? 16 : 0)
// ---------------------------------------------------------------------------
__global__ void packw_kernel(const float* __restrict__ w, __hip_bfloat16* __restrict__ wp) {
    int t = blockIdx.x * 256 + threadIdx.x;          // 65536
    int j = t & 15, lane = (t >> 4) & 31;
    int gidx = t >> 9;                               // kstep*16 + ntile
    int ntile = gidx & 15, kstep = gidx >> 4;
    int n = ntile * 16 + (lane & 15);
    int k = kstep * 32 + (lane >> 4) * 8 + (j & 7) + (j >> 3) * 16;
    wp[t] = __float2bfloat16(w[k * 256 + n]);
}

// ---------------------------------------------------------------------------
// 8) WMMA GEMM: (M x 256) bf16 @ (256 x 256) bf16 -> (M x 256) f32 + bias.
//    B panels double-buffered in LDS via async-to-LDS (ASYNCcnt path).
//    B fragments consumed in register chunks of 8 so one DS wait covers an
//    8-WMMA burst and the next chunk's ds_loads overlap the matrix pipe.
// ---------------------------------------------------------------------------
__device__ __forceinline__ void stage_b(const __hip_bfloat16* __restrict__ src,
                                        __hip_bfloat16* dst, int tid) {
#if USE_ASYNC_LDS
#pragma unroll
    for (int i = 0; i < 4; ++i) {
        __builtin_amdgcn_global_load_async_to_lds_b128(
            (GLOBAL_AS async_i4*)(src + (size_t)(tid * 4 + i) * 8),
            (LDS_AS async_i4*)(dst + (size_t)(tid * 4 + i) * 8), 0, 0);
    }
#else
    const uint4* s = (const uint4*)src;
    uint4* d = (uint4*)dst;
#pragma unroll
    for (int i = 0; i < 4; ++i) d[tid * 4 + i] = s[tid * 4 + i];
#endif
}

__global__ void __launch_bounds__(256)
gemm_wmma_kernel(const __hip_bfloat16* __restrict__ A,
                 const __hip_bfloat16* __restrict__ Wp,
                 const float* __restrict__ bias, float* __restrict__ out) {
    __shared__ __hip_bfloat16 Bs[2][16 * 32 * 16];   // 2 x 16 KB panels
    int tid = threadIdx.x;
    int waveId = tid >> 5;
    int lane   = tid & 31;
    int g = lane >> 4;
    int mbase = blockIdx.x * 128 + waveId * 16;
    const __hip_bfloat16* arow = A + (size_t)(mbase + (lane & 15)) * 256;

    stage_b(Wp, Bs[0], tid);                          // k-step 0 panel
#if USE_ASYNC_LDS
    __builtin_amdgcn_s_wait_asynccnt(0);
#endif
    __syncthreads();

    v8f acc[16] = {};
    for (int kstep = 0; kstep < 8; ++kstep) {
        int buf = kstep & 1;
        if (kstep < 7)                                // prefetch next panel
            stage_b(Wp + (size_t)(kstep + 1) * 8192, Bs[buf ^ 1], tid);

        // A fragment (ISA 16-bit layout): k = g*8 + (j%8) + (j>=8 ? 16 : 0)
        int k0 = kstep * 32 + g * 8;
        v8bf alo = *(const v8bf*)(arow + k0);
        v8bf ahi = *(const v8bf*)(arow + k0 + 16);
        if (kstep < 7) __builtin_prefetch(arow + k0 + 32, 0, 3);
        v16bf a;
#pragma unroll
        for (int j = 0; j < 8; ++j) { a[j] = alo[j]; a[8 + j] = ahi[j]; }

        const __hip_bfloat16* bbase = Bs[buf] + (size_t)lane * 16;
#pragma unroll
        for (int half = 0; half < 2; ++half) {
            v16bf bf[8];
#pragma unroll
            for (int n = 0; n < 8; ++n)
                bf[n] = *(const v16bf*)(bbase + (size_t)(half * 8 + n) * 512);
#pragma unroll
            for (int n = 0; n < 8; ++n)
                acc[half * 8 + n] = __builtin_amdgcn_wmma_f32_16x16x32_bf16(
                    false, a, false, bf[n], (short)0, acc[half * 8 + n], false, false);
        }
#if USE_ASYNC_LDS
        if (kstep < 7) __builtin_amdgcn_s_wait_asynccnt(0);
#endif
        __syncthreads();
    }

    // D layout: VGPR r, lanes 0-15 -> M = r, lanes 16-31 -> M = 8+r; N = lane%16
#pragma unroll
    for (int n = 0; n < 16; ++n) {
        int col = n * 16 + (lane & 15);
        float bv = bias[col];
#pragma unroll
        for (int r = 0; r < 8; ++r) {
            int m = mbase + g * 8 + r;
            out[(size_t)m * 256 + col] = acc[n][r] + bv;
        }
    }
}

// ---------------------------------------------------------------------------
// 9) Fused BN+ReLU+pool over nsample=32: max and mean per (row, channel).
// ---------------------------------------------------------------------------
__global__ void __launch_bounds__(256)
pool_kernel(const float* __restrict__ pre, const float* __restrict__ sum,
            const float* __restrict__ sq, const float* __restrict__ gamma,
            const float* __restrict__ beta, const float* __restrict__ new_xyz,
            float* __restrict__ l1out, float* __restrict__ x_in) {
    int row = blockIdx.x;            // 0..8191
    int c   = threadIdx.x;           // 0..255
    const float invN = 1.0f / 262144.0f;
    float m  = sum[c] * invN;
    float v  = sq[c] * invN - m * m;
    float sc = gamma[c] * rsqrtf(v + BN_EPS);
    float sh = beta[c] - m * sc;
    const float* src = pre + ((size_t)row * NSAMPLE) * 256 + c;
    float mx = -1e30f, mn = 0.f;
#pragma unroll 4
    for (int k = 0; k < NSAMPLE; ++k) {
        float x = fmaxf(src[(size_t)k * 256] * sc + sh, 0.f);
        mx = fmaxf(mx, x);
        mn += x;
    }
    mn *= (1.0f / NSAMPLE);
    l1out[(size_t)row * 512 + c]        = mx;
    l1out[(size_t)row * 512 + 256 + c]  = mn;
    x_in [(size_t)row * 515 + 3 + c]       = mx;
    x_in [(size_t)row * 515 + 3 + 256 + c] = mn;
    if (c < 3) x_in[(size_t)row * 515 + c] = new_xyz[(size_t)row * 3 + c];
}

// ---------------------------------------------------------------------------
// 10) FC1: (8192 x 515) @ (515 x 128) + b. One block per row, LDS row stage.
// ---------------------------------------------------------------------------
__global__ void __launch_bounds__(128)
fc1_kernel(const float* __restrict__ x, const float* __restrict__ w,
           const float* __restrict__ b, float* __restrict__ out) {
    __shared__ float xs[515];
    int row = blockIdx.x, c = threadIdx.x;
    for (int i = c; i < 515; i += 128) xs[i] = x[(size_t)row * 515 + i];
    __syncthreads();
    float acc = b[c];
    for (int k = 0; k < 515; ++k) acc += xs[k] * w[(size_t)k * 128 + c];
    out[(size_t)row * 128 + c] = acc;
}

// ---------------------------------------------------------------------------
// 11) FC2 + log_softmax: one wave32 per row; lanes cover classes {lane, lane+32}.
// ---------------------------------------------------------------------------
__global__ void __launch_bounds__(256)
fc2_kernel(const float* __restrict__ act, const float* __restrict__ w,
           const float* __restrict__ b, float* __restrict__ out) {
    int wave = (blockIdx.x * 256 + threadIdx.x) >> 5;   // 0..8191
    int lane = threadIdx.x & 31;
    const float* a = act + (size_t)wave * 128;
    int c0 = lane, c1 = lane + 32;
    float l0 = b[c0], l1 = 0.f;
    for (int k = 0; k < 128; ++k) l0 += a[k] * w[k * 40 + c0];
    bool has1 = (c1 < 40);
    if (has1) {
        l1 = b[c1];
        for (int k = 0; k < 128; ++k) l1 += a[k] * w[k * 40 + c1];
    }
    float mx = fmaxf(l0, has1 ? l1 : -1e30f);
    for (int off = 16; off > 0; off >>= 1) mx = fmaxf(mx, __shfl_xor(mx, off, 32));
    float e = __expf(l0 - mx) + (has1 ? __expf(l1 - mx) : 0.f);
    for (int off = 16; off > 0; off >>= 1) e += __shfl_xor(e, off, 32);
    float lse = __logf(e);
    out[(size_t)wave * 40 + c0] = l0 - mx - lse;
    if (has1) out[(size_t)wave * 40 + c1] = l1 - mx - lse;
}

// ---------------------------------------------------------------------------
// host launcher
// ---------------------------------------------------------------------------
extern "C" void kernel_launch(void* const* d_in, const int* in_sizes, int n_in,
                              void* d_out, int out_size, void* d_ws, size_t ws_size,
                              hipStream_t stream) {
    const float* xyz   = (const float*)d_in[0];
    const float* w1    = (const float*)d_in[1];
    const float* b1    = (const float*)d_in[2];
    const float* g1    = (const float*)d_in[3];
    const float* be1   = (const float*)d_in[4];
    const float* w2    = (const float*)d_in[5];
    const float* b2    = (const float*)d_in[6];
    const float* g2    = (const float*)d_in[7];
    const float* be2   = (const float*)d_in[8];
    const float* w3    = (const float*)d_in[9];
    const float* b3    = (const float*)d_in[10];
    const float* g3    = (const float*)d_in[11];
    const float* be3   = (const float*)d_in[12];
    const float* fc1_w = (const float*)d_in[13];
    const float* fc1_b = (const float*)d_in[14];
    const float* bn1_g = (const float*)d_in[15];
    const float* bn1_b = (const float*)d_in[16];
    const float* fc2_w = (const float*)d_in[17];
    const float* fc2_b = (const float*)d_in[18];

    char* ws = (char*)d_ws;
    const size_t OFF_P      = 0;                              // 16*4096*3 f32
    const size_t OFF_NORM   = OFF_P      + 786432;
    const size_t OFF_IDX    = OFF_NORM   + 786432;            // 8192 i32
    const size_t OFF_NEWXYZ = OFF_IDX    + 32768;             // 8192*3 f32
    const size_t OFF_H0     = OFF_NEWXYZ + 98304;             // 262144*6 f32
    const size_t OFF_PRE    = OFF_H0     + 6291456;           // 262144*256 f32
    const size_t OFF_ACT    = OFF_PRE    + 268435456;         // 262144*256 bf16
    const size_t OFF_WPACK  = OFF_ACT    + 134217728;         // 65536 bf16
    const size_t OFF_SUM    = OFF_WPACK  + 131072;            // 256 f32
    const size_t OFF_SQ     = OFF_SUM    + 1024;              // 256 f32
    const size_t OFF_PSUM   = OFF_SQ     + 1024;              // 256*256 f32
    const size_t OFF_PSQ    = OFF_PSUM   + 262144;            // 256*256 f32
    const size_t OFF_XIN    = OFF_PSQ    + 262144;            // 8192*515 f32
    const size_t OFF_FC1PRE = OFF_XIN    + 16875520;          // 8192*128 f32
    const size_t OFF_FC1ACT = OFF_FC1PRE + 4194304;           // 8192*128 f32

    float* p       = (float*)(ws + OFF_P);
    float* nrm     = (float*)(ws + OFF_NORM);
    int*   idx     = (int*)  (ws + OFF_IDX);
    float* new_xyz = (float*)(ws + OFF_NEWXYZ);
    float* h0      = (float*)(ws + OFF_H0);
    float* pre     = (float*)(ws + OFF_PRE);
    __hip_bfloat16* act   = (__hip_bfloat16*)(ws + OFF_ACT);
    __hip_bfloat16* wpack = (__hip_bfloat16*)(ws + OFF_WPACK);
    float* ssum    = (float*)(ws + OFF_SUM);
    float* ssq     = (float*)(ws + OFF_SQ);
    float* psum    = (float*)(ws + OFF_PSUM);
    float* psq     = (float*)(ws + OFF_PSQ);
    float* x_in    = (float*)(ws + OFF_XIN);
    float* fc1_pre = (float*)(ws + OFF_FC1PRE);
    float* fc1_act = (float*)(ws + OFF_FC1ACT);

    float* out_x  = (float*)d_out;            // (16,512,40)
    float* out_l1 = (float*)d_out + 327680;   // (16,512,512)

    const long long ROWS = 262144LL;          // 16*512*32
    const float invN_bn = 1.0f / 262144.0f;
    const float invN_fc = 1.0f / 8192.0f;

    extract_kernel<<<256, 256, 0, stream>>>(xyz, p, nrm);
    fps_kernel<<<BATCH, 256, 0, stream>>>(p, idx);
    ballquery_kernel<<<1024, 256, 0, stream>>>(p, nrm, idx, new_xyz, h0);

    // ---- layer 1 (FMA, K=6) ----
    gemm1_kernel<<<262144, 256, 0, stream>>>(h0, w1, b1, pre);
    stats_partial_kernel<<<256, 256, 0, stream>>>(pre, ROWS, 256, 256, psum, psq);
    stats_final_kernel<<<1, 256, 0, stream>>>(psum, psq, 256, 256, ssum, ssq);
    bnrelu_kernel<true><<<262144, 256, 0, stream>>>(pre, ssum, ssq, g1, be1, invN_bn,
                                                    act, ROWS * 256, 255);
    // ---- layer 2 (WMMA bf16) ----
    packw_kernel<<<256, 256, 0, stream>>>(w2, wpack);
    gemm_wmma_kernel<<<2048, 256, 0, stream>>>(act, wpack, b2, pre);
    stats_partial_kernel<<<256, 256, 0, stream>>>(pre, ROWS, 256, 256, psum, psq);
    stats_final_kernel<<<1, 256, 0, stream>>>(psum, psq, 256, 256, ssum, ssq);
    bnrelu_kernel<true><<<262144, 256, 0, stream>>>(pre, ssum, ssq, g2, be2, invN_bn,
                                                    act, ROWS * 256, 255);
    // ---- layer 3 (WMMA bf16) ----
    packw_kernel<<<256, 256, 0, stream>>>(w3, wpack);
    gemm_wmma_kernel<<<2048, 256, 0, stream>>>(act, wpack, b3, pre);
    stats_partial_kernel<<<256, 256, 0, stream>>>(pre, ROWS, 256, 256, psum, psq);
    stats_final_kernel<<<1, 256, 0, stream>>>(psum, psq, 256, 256, ssum, ssq);

    // ---- BN3 + ReLU + max/mean pool, build x_in ----
    pool_kernel<<<8192, 256, 0, stream>>>(pre, ssum, ssq, g3, be3, new_xyz, out_l1, x_in);

    // ---- FC1 + BN + ReLU ----
    fc1_kernel<<<8192, 128, 0, stream>>>(x_in, fc1_w, fc1_b, fc1_pre);
    stats_partial_kernel<<<64, 128, 0, stream>>>(fc1_pre, 8192LL, 128, 64, psum, psq);
    stats_final_kernel<<<1, 128, 0, stream>>>(psum, psq, 128, 64, ssum, ssq);
    bnrelu_kernel<false><<<4096, 256, 0, stream>>>(fc1_pre, ssum, ssq, bn1_g, bn1_b, invN_fc,
                                                   fc1_act, 8192LL * 128, 127);

    // ---- FC2 + log_softmax ----
    fc2_kernel<<<1024, 256, 0, stream>>>(fc1_act, fc2_w, fc2_b, out_x);
}
